// CausalMaskedLinear_24550033064450
// MI455X (gfx1250) — matrix operand order repaired and checked by reading, not compiled, not run on hardware
//
#include <hip/hip_runtime.h>

typedef __attribute__((ext_vector_type(16))) __bf16 v16bf;
typedef __attribute__((ext_vector_type(8)))  __bf16 v8bf;
typedef __attribute__((ext_vector_type(8)))  float  v8f;

// Problem constants (match the reference).
constexpr int TS       = 32;          // NUM_TIME_STEPS
constexpr int F        = 128;         // IN_FEAT == OUT_FEAT
constexpr int IN_SIZE  = TS * F;      // 4096
constexpr int TRI      = 8;           // TRI_BLOCK
constexpr int TILE_M   = 128;         // batch rows per workgroup
constexpr int S        = F + 8;       // LDS row stride in bf16 elems (272B, 16B-aligned, bank-staggered)

__device__ __forceinline__ v16bf load_frag(const __bf16* p0, const __bf16* p1) {
    const v8bf lo = *(const v8bf*)p0;
    const v8bf hi = *(const v8bf*)p1;
    v16bf f;
    #pragma unroll
    for (int i = 0; i < 8; ++i) { f[i] = lo[i]; f[i + 8] = hi[i]; }
    return f;
}

__global__ __launch_bounds__(256)
void cml_wmma_kernel(const float* __restrict__ x,
                     const float* __restrict__ w,
                     const float* __restrict__ bias,
                     float* __restrict__ y)
{
    __shared__ __bf16 Xs[TILE_M * S];   // 128 batch rows  x 128 k   (bf16)
    __shared__ __bf16 Ws[F * S];        // 128 out feats   x 128 k   (bf16)

    const int tid  = threadIdx.x;
    const int lane = tid & 31;
    const int wv   = tid >> 5;                 // 8 waves
    const int t    = blockIdx.x;               // output time block (fast-varying: x slab stays in L2)
    const int row0 = blockIdx.y * TILE_M;      // batch tile
    const int lo   = (t >= TRI) ? (t - TRI + 1) : 0;

    // 4x2 wave grid: each wave owns 32 rows (2 M-tiles) x 64 cols (4 N-tiles).
    const int m0   = (wv >> 1) * 32;
    const int n0   = (wv & 1) * 64;
    const int lrow = lane & 15;
    const int lhi  = lane >> 4;                // K-half selector (contiguous k = 16*lhi + s, this code's order)

    v8f acc[2][4];
    #pragma unroll
    for (int mt = 0; mt < 2; ++mt)
        #pragma unroll
        for (int nt = 0; nt < 4; ++nt)
            #pragma unroll
            for (int j = 0; j < 8; ++j) acc[mt][nt][j] = 0.0f;

    for (int kb = lo; kb <= t; ++kb) {
        const int kbase = kb * F;

        // Prefetch next K-block (global_prefetch_b8; overlaps with this block's math).
        if (kb < t) {
            const int pr = tid >> 1;                    // 2 threads/row, 256B each
            const int pc = (tid & 1) * 64;
            __builtin_prefetch(x + (size_t)(row0 + pr) * IN_SIZE + (kbase + F) + pc, 0, 0);
            __builtin_prefetch(w + (size_t)(t * F + pr) * IN_SIZE + (kbase + F) + pc, 0, 0);
        }

        // ---- Stage X tile: 128x128 fp32 -> bf16 in LDS ----
        // 128 rows * 32 float4 = 4096 slots, 256 threads * 16 iters.
        #pragma unroll
        for (int it = 0; it < 16; ++it) {
            const int slot = tid + it * 256;
            const int r    = slot >> 5;
            const int c    = (slot & 31) << 2;
            const float4 v = *(const float4*)(x + (size_t)(row0 + r) * IN_SIZE + kbase + c);
            __bf16* dst = &Xs[r * S + c];
            dst[0] = (__bf16)v.x; dst[1] = (__bf16)v.y;
            dst[2] = (__bf16)v.z; dst[3] = (__bf16)v.w;
        }
        // ---- Stage W tile: 128x128 fp32 -> bf16 in LDS ----
        #pragma unroll
        for (int it = 0; it < 16; ++it) {
            const int slot = tid + it * 256;
            const int r    = slot >> 5;
            const int c    = (slot & 31) << 2;
            const float4 v = *(const float4*)(w + (size_t)(t * F + r) * IN_SIZE + kbase + c);
            __bf16* dst = &Ws[r * S + c];
            dst[0] = (__bf16)v.x; dst[1] = (__bf16)v.y;
            dst[2] = (__bf16)v.z; dst[3] = (__bf16)v.w;
        }
        __syncthreads();

        // ---- 4 K-steps of 32 per staged block ----
        #pragma unroll
        for (int kk = 0; kk < 4; ++kk) {
            const int k0 = kk * 32;

            // A fragments (16x32 bf16): lane holds row m; K-half by lhi.
            const int ka = k0 + lhi * 16;
            const v16bf a0 = load_frag(&Xs[(m0      + lrow) * S + ka],
                                       &Xs[(m0      + lrow) * S + ka + 8]);
            const v16bf a1 = load_frag(&Xs[(m0 + 16 + lrow) * S + ka],
                                       &Xs[(m0 + 16 + lrow) * S + ka + 8]);

            // B fragments (32x16 bf16): lane holds col n; k = k0 + lhi*16 .. +15.
            const int kc = k0 + lhi * 16;
            const v16bf b0 = load_frag(&Ws[(n0      + lrow) * S + kc],
                                       &Ws[(n0      + lrow) * S + kc + 8]);
            const v16bf b1 = load_frag(&Ws[(n0 + 16 + lrow) * S + kc],
                                       &Ws[(n0 + 16 + lrow) * S + kc + 8]);
            const v16bf b2 = load_frag(&Ws[(n0 + 32 + lrow) * S + kc],
                                       &Ws[(n0 + 32 + lrow) * S + kc + 8]);
            const v16bf b3 = load_frag(&Ws[(n0 + 48 + lrow) * S + kc],
                                       &Ws[(n0 + 48 + lrow) * S + kc + 8]);

            acc[0][0] = __builtin_amdgcn_wmma_f32_16x16x32_bf16(false, a0, false, b0, (short)0, acc[0][0], false, false);
            acc[0][1] = __builtin_amdgcn_wmma_f32_16x16x32_bf16(false, a0, false, b1, (short)0, acc[0][1], false, false);
            acc[0][2] = __builtin_amdgcn_wmma_f32_16x16x32_bf16(false, a0, false, b2, (short)0, acc[0][2], false, false);
            acc[0][3] = __builtin_amdgcn_wmma_f32_16x16x32_bf16(false, a0, false, b3, (short)0, acc[0][3], false, false);
            acc[1][0] = __builtin_amdgcn_wmma_f32_16x16x32_bf16(false, a1, false, b0, (short)0, acc[1][0], false, false);
            acc[1][1] = __builtin_amdgcn_wmma_f32_16x16x32_bf16(false, a1, false, b1, (short)0, acc[1][1], false, false);
            acc[1][2] = __builtin_amdgcn_wmma_f32_16x16x32_bf16(false, a1, false, b2, (short)0, acc[1][2], false, false);
            acc[1][3] = __builtin_amdgcn_wmma_f32_16x16x32_bf16(false, a1, false, b3, (short)0, acc[1][3], false, false);
        }
        __syncthreads();
    }

    // ---- Epilogue: D layout (16x16 f32): lane n = lrow; VGPR r -> M = lhi*8 + r ----
    const int colbase = t * F + n0;
    #pragma unroll
    for (int nt = 0; nt < 4; ++nt) {
        const int col = colbase + nt * 16 + lrow;
        const float bv = bias[col];
        #pragma unroll
        for (int mt = 0; mt < 2; ++mt) {
            #pragma unroll
            for (int r = 0; r < 8; ++r) {
                const int rowm = row0 + m0 + mt * 16 + lhi * 8 + r;
                y[(size_t)rowm * IN_SIZE + col] = acc[mt][nt][r] + bv;
            }
        }
    }
}

extern "C" void kernel_launch(void* const* d_in, const int* in_sizes, int n_in,
                              void* d_out, int out_size, void* d_ws, size_t ws_size,
                              hipStream_t stream) {
    const float* x    = (const float*)d_in[0];
    const float* w    = (const float*)d_in[1];
    const float* bias = (const float*)d_in[2];
    // d_in[3] is the mask; its block-band structure (TRI_BLOCK=8) is baked into the kernel.
    float* y = (float*)d_out;

    const int batch = in_sizes[0] / IN_SIZE;   // 8192
    dim3 grid(TS, batch / TILE_M);             // (32, 64): t fast-varying for L2 reuse
    cml_wmma_kernel<<<grid, 256, 0, stream>>>(x, w, bias, y);
}
